// Classifier_11665131176106
// MI455X (gfx1250) — compile-verified, matrix operand order checked by reading, not run on hardware
//
#include <hip/hip_runtime.h>
#include <hip/hip_bf16.h>

typedef __attribute__((ext_vector_type(16))) _Float16 v16h;
typedef __attribute__((ext_vector_type(8)))  _Float16 v8h;
typedef __attribute__((ext_vector_type(8)))  float    v8f;
typedef __attribute__((ext_vector_type(4)))  unsigned int u32x4;
typedef __attribute__((ext_vector_type(8)))  int      i32x8;
typedef __attribute__((ext_vector_type(4)))  int      i32x4;

#define BATCH 1024
#define NFWD 4

// ---------------------------------------------------------------------------
// Tensor Data Mover staging (CDNA5): one wave issues a 1-row-tile descriptor
// for the contiguous weight-tile global->LDS copy; TENSORcnt guards completion.
// Arity hedged between ROCm 7.2 (5-arg) and therock clang-23 (6-arg).
// ---------------------------------------------------------------------------
#if __has_builtin(__builtin_amdgcn_tensor_load_to_lds)
#define CDNA5_HAS_TDM 1
#if __has_include(<hip/amd_detail/amd_gfx1250_TDM.h>)
#define TDM_CALL(g0, g1, za, zb) \
    __builtin_amdgcn_tensor_load_to_lds((g0), (g1), (za), (zb), (i32x8)(0), 0)
#else
#define TDM_CALL(g0, g1, za, zb) \
    __builtin_amdgcn_tensor_load_to_lds((g0), (g1), (za), (zb), 0)
#endif
#else
#define CDNA5_HAS_TDM 0
#endif

__device__ __forceinline__ void wait_tensorcnt0() {
#if __has_builtin(__builtin_amdgcn_s_wait_tensorcnt)
    __builtin_amdgcn_s_wait_tensorcnt(0);
#else
    asm volatile("s_wait_tensorcnt 0x0" ::: "memory");
#endif
}

#if CDNA5_HAS_TDM
// Contiguous 1-D global->LDS DMA of nbytes (multiple of 4) via TDM.
__device__ __forceinline__ void tdm_load_1d(const void* gsrc, void* lds_dst, int nbytes) {
    const unsigned lds = (unsigned)(unsigned long long)lds_dst;        // low 32b = LDS offset
    const unsigned long long ga = (unsigned long long)gsrc;
    const int nd = nbytes >> 2;                                        // dwords
    u32x4 g0;
    g0[0] = 1u;                                                        // count=1, user desc
    g0[1] = lds;                                                       // lds_addr
    g0[2] = (unsigned)(ga & 0xffffffffull);                            // global_addr[31:0]
    g0[3] = (unsigned)((ga >> 32) & 0x1ffffffull) | (2u << 30);        // ga[56:32] | type=2
    i32x8 g1;
    g1[0] = 0x20000;                                                   // data_size=4B, mask=0
    g1[1] = (int)((nd & 0xffff) << 16);                                // tensor_dim0[15:0]
    g1[2] = (int)(((unsigned)nd >> 16) | (1u << 16));                  // dim0[31:16] | dim1.lo=1
    g1[3] = (int)((unsigned)nd << 16);                                 // dim1.hi=0 | tile_dim0
    g1[4] = 1;                                                         // tile_dim1=1, tile_dim2=0
    g1[5] = nd;                                                        // tensor_dim0_stride lo32
    g1[6] = 0;
    g1[7] = 0;
    i32x4 z;
    z[0] = 0; z[1] = 0; z[2] = 0; z[3] = 0;
    TDM_CALL(g0, g1, z, z);
}
#endif

// ---------------------------------------------------------------------------
// small helpers
// ---------------------------------------------------------------------------
__device__ __forceinline__ unsigned int mix_hash(unsigned int a, unsigned int b) {
    unsigned int s = a * 0x9E3779B9u ^ (b + 0x85EBCA6Bu);
    s ^= s >> 16; s *= 0x7feb352du;
    s ^= s >> 15; s *= 0x846ca68bu;
    s ^= s >> 16;
    return s;
}

__device__ __forceinline__ v8f v8f_zero() {
    v8f z;
#pragma unroll
    for (int e = 0; e < 8; ++e) z[e] = 0.0f;
    return z;
}

// 16-bit WMMA A/B fragment gather from an LDS row (wave32 striping):
// lanes 0-15: K = e + (e>=8 ? 8 : 0); lanes 16-31: +8.
__device__ __forceinline__ v16h load_frag_lds(const _Float16* base, int half) {
    const v8h* lo = (const v8h*)(base + half * 8);
    const v8h* hi = (const v8h*)(base + 16 + half * 8);
    v8h l = *lo, h = *hi;
    v16h r;
#pragma unroll
    for (int e = 0; e < 8; ++e) { r[e] = l[e]; r[8 + e] = h[e]; }
    return r;
}

// ---------------------------------------------------------------------------
// weight repack: OIHW f32 -> [(K/32)][Cout][32] f16  (K = tap*Cin + c)
// ---------------------------------------------------------------------------
__global__ void repack_conv3x3(const float* __restrict__ w, _Float16* __restrict__ wkT,
                               int Cin, int Cout) {
    long total = (long)Cin * 9 * Cout;
    long idx = (long)blockIdx.x * 256 + threadIdx.x;
    if (idx >= total) return;
    int k = (int)(idx / Cout);
    int n = (int)(idx - (long)k * Cout);
    int tap = k / Cin;
    int c   = k - tap * Cin;
    float v = w[((long)n * Cin + c) * 9 + tap];
    wkT[(((long)(k >> 5)) * Cout + n) * 32 + (k & 31)] = (_Float16)v;
}

__global__ void repack_conv1x1(const float* __restrict__ w, _Float16* __restrict__ wcT,
                               int Cin, int Cout) {
    long total = (long)Cin * Cout;
    long idx = (long)blockIdx.x * 256 + threadIdx.x;
    if (idx >= total) return;
    int c = (int)(idx / Cout);
    int n = (int)(idx - (long)c * Cout);
    float v = w[(long)n * Cin + c];
    wcT[(((long)(c >> 5)) * Cout + n) * 32 + (c & 31)] = (_Float16)v;
}

// ---------------------------------------------------------------------------
// dropout + conv0 (1 -> 32 channels, 1x1), output fp16 NHWC
// ---------------------------------------------------------------------------
__global__ void drop_conv0(const float* __restrict__ x,
                           const float* __restrict__ w0,
                           const float* __restrict__ b0,
                           _Float16* __restrict__ out, int f) {
    long total = (long)BATCH * 28 * 28;
    long p = (long)blockIdx.x * 256 + threadIdx.x;
    if (p >= total) return;
    unsigned int s = mix_hash((unsigned int)p, (unsigned int)f);
    float val = (s & 1u) ? (x[p] * 2.0f) : 0.0f;   // keep prob 0.5, scale 1/(1-p)
    _Float16* o = out + p * 32;
#pragma unroll 8
    for (int c = 0; c < 32; ++c)
        o[c] = (_Float16)(w0[c] * val + b0[c]);
}

// ---------------------------------------------------------------------------
// fused block: implicit-GEMM 3x3 conv + 1x1 coupling conv + bias + ReLU +
// ReZero scaling, via v_wmma_f32_16x16x32_f16. Weight tiles staged with the
// Tensor Data Mover when available. NT = Cout/16 (templated: register accs).
// ---------------------------------------------------------------------------
template <int NT>
__global__ __launch_bounds__(256)
void conv_block_gemm(const _Float16* __restrict__ in,   // [B,H,W,Cin] f16
                     const _Float16* __restrict__ wkT,  // [(9*Cin/32)][Cout][32]
                     const _Float16* __restrict__ wcT,  // [(Cin/32)][Cout][32]
                     const float* __restrict__ bconv,
                     const float* __restrict__ bcoup,
                     const float* __restrict__ rzp,
                     _Float16* __restrict__ out,        // [B,H,W,Cout] f16
                     int Cin, int H, int W) {
    constexpr int Cout = NT * 16;
    __shared__ __align__(16) _Float16 As[128 * 32];
    __shared__ __align__(16) _Float16 Bs[Cout * 32];
    __shared__ __align__(16) _Float16 Cs[Cout * 32];

    const int tid  = threadIdx.x;
    const int lane = tid & 31;
    const int wave = tid >> 5;          // 0..7 -> M sub-tile
    const int half = (lane >> 4) & 1;
    const int ln   = lane & 15;

    const int HW   = H * W;
    const long Mtot = (long)BATCH * HW;
    const long tile0 = (long)blockIdx.x * 128;

    v8f accC[NT], accD[NT];
#pragma unroll
    for (int j = 0; j < NT; ++j) { accC[j] = v8f_zero(); accD[j] = v8f_zero(); }

    // A-staging assignment: thread t -> row r = t/2, 16-channel group g = t%2
    const int r = tid >> 1;
    const int g = tid & 1;
    long p = tile0 + r;
    int py = 0, px = 0;
    long pixBase = 0;
    const bool prow = (p < Mtot);
    if (prow) {
        int pb = (int)(p / HW);
        int rem = (int)(p - (long)pb * HW);
        py = rem / W;
        px = rem - py * W;
        pixBase = (long)pb * HW;
    }

    const int chunksPerTap = Cin >> 5;
    const int nChunks = chunksPerTap * 9;
    const int bBytes = Cout * 32 * 2;   // B tile bytes
    const int bCopy  = bBytes / 16;     // uint4 elements

    for (int kk = 0; kk < nChunks; ++kk) {
        const int tap = kk / chunksPerTap;
        const int cb  = (kk - tap * chunksPerTap) << 5;
        const int ky  = tap / 3 - 1;
        const int kx  = tap % 3 - 1;

        // ---- stage A tile: 128 rows x 32 K (fp16), 32 B per thread ----
        {
            const int yy = py + ky, xx = px + kx;
            const bool valid = prow && yy >= 0 && yy < H && xx >= 0 && xx < W;
            uint4 z0{}, z1{};
            if (valid) {
                const uint4* src = (const uint4*)(in +
                    ((pixBase + (long)yy * W + xx) * Cin + cb + g * 16));
                z0 = src[0];
                z1 = src[1];
            }
            uint4* dst = (uint4*)(&As[r * 32 + g * 16]);
            dst[0] = z0;
            dst[1] = z1;
        }
#if CDNA5_HAS_TDM
        // ---- stage B (and coupling) tiles via the Tensor Data Mover ----
        if (tid < 32) {                 // wave 0 issues the DMA descriptors
            tdm_load_1d(wkT + (long)kk * Cout * 32, &Bs[0], bBytes);
            if (tap == 4)
                tdm_load_1d(wcT + (long)(cb >> 5) * Cout * 32, &Cs[0], bBytes);
            wait_tensorcnt0();
        }
#else
        // ---- fallback: cooperative contiguous copy ----
        {
            const uint4* src = (const uint4*)(wkT + (long)kk * Cout * 32);
            uint4* dst = (uint4*)Bs;
            for (int e = tid; e < bCopy; e += 256) dst[e] = src[e];
        }
        if (tap == 4) {
            const uint4* src = (const uint4*)(wcT + (long)(cb >> 5) * Cout * 32);
            uint4* dst = (uint4*)Cs;
            for (int e = tid; e < bCopy; e += 256) dst[e] = src[e];
        }
#endif
        // prefetch next chunk's weights into near cache while we compute
        if (kk + 1 < nChunks)
            __builtin_prefetch((const void*)(wkT + (long)(kk + 1) * Cout * 32), 0, 3);

        __syncthreads();

        // ---- compute: 1 A fragment per wave, NT B fragments ----
        const int m = wave * 16 + ln;
        const v16h a = load_frag_lds(&As[m * 32], half);
#pragma unroll
        for (int j = 0; j < NT; ++j) {
            const v16h b = load_frag_lds(&Bs[(j * 16 + ln) * 32], half);
            accC[j] = __builtin_amdgcn_wmma_f32_16x16x32_f16(
                false, a, false, b, (short)0, accC[j], false, false);
        }
        if (tap == 4) {
#pragma unroll
            for (int j = 0; j < NT; ++j) {
                const v16h b = load_frag_lds(&Cs[(j * 16 + ln) * 32], half);
                accD[j] = __builtin_amdgcn_wmma_f32_16x16x32_f16(
                    false, a, false, b, (short)0, accD[j], false, false);
            }
        }
        __syncthreads();
    }

    // ---- epilogue: rz*relu(conv+b) + rz*(coup+cb), fp16 NHWC out ----
    const float rz = *rzp;
#pragma unroll
    for (int j = 0; j < NT; ++j) {
        const int n = j * 16 + ln;
        const float bc = bconv[n];
        const float bd = bcoup[n];
#pragma unroll
        for (int e = 0; e < 8; ++e) {
            const int mm = e + half * 8;                 // D: M = vgpr + half*8
            const long row = tile0 + (long)wave * 16 + mm;
            if (row < Mtot) {
                float vc = accC[j][e] + bc;
                vc = vc > 0.0f ? vc : 0.0f;
                const float val = rz * vc + rz * (accD[j][e] + bd);
                out[row * Cout + n] = (_Float16)val;
            }
        }
    }
}

// ---------------------------------------------------------------------------
// deterministic fractional max pool (kernel 2, evenly spaced starts), NHWC f16
// ---------------------------------------------------------------------------
__global__ void fpool_kernel(const _Float16* __restrict__ in, _Float16* __restrict__ out,
                             int Hin, int Hout, int C) {
    long total = (long)BATCH * Hout * Hout * C;
    long idx = (long)blockIdx.x * 256 + threadIdx.x;
    if (idx >= total) return;
    int c = (int)(idx % C);
    long t = idx / C;
    int xo = (int)(t % Hout); t /= Hout;
    int yo = (int)(t % Hout);
    int b  = (int)(t / Hout);
    const int sy = (yo * (Hin - 2)) / (Hout - 1);
    const int sx = (xo * (Hin - 2)) / (Hout - 1);
    const long base = ((long)b * Hin + sy) * Hin + sx;
    float m0 = (float)in[(base)           * C + c];
    float m1 = (float)in[(base + 1)       * C + c];
    float m2 = (float)in[(base + Hin)     * C + c];
    float m3 = (float)in[(base + Hin + 1) * C + c];
    float m = m0 > m1 ? m0 : m1;
    m = m > m2 ? m : m2;
    m = m > m3 ? m : m3;
    out[idx] = (_Float16)m;
}

// ---------------------------------------------------------------------------
// final 3x3 conv (192->1, pad 1, 4x4) + flatten(16) + linear(16->10)
// ---------------------------------------------------------------------------
__global__ void final_linear(const _Float16* __restrict__ act,  // [B,4,4,192] f16
                             const float* __restrict__ fw,      // [1,192,3,3]
                             const float* __restrict__ fb,      // [1]
                             const float* __restrict__ lw,      // [10,16]
                             const float* __restrict__ lb,      // [10]
                             float* __restrict__ scores, int f) {
    int b = blockIdx.x * 64 + threadIdx.x;
    if (b >= BATCH) return;
    float t[16];
#pragma unroll
    for (int yo = 0; yo < 4; ++yo)
#pragma unroll
        for (int xo = 0; xo < 4; ++xo) {
            float acc = fb[0];
#pragma unroll
            for (int ky = -1; ky <= 1; ++ky)
#pragma unroll
                for (int kx = -1; kx <= 1; ++kx) {
                    const int yy = yo + ky, xx = xo + kx;
                    if (yy < 0 || yy >= 4 || xx < 0 || xx >= 4) continue;
                    const int tap = (ky + 1) * 3 + (kx + 1);
                    const _Float16* row = act + (((long)b * 16 + yy * 4 + xx) * 192);
                    for (int c = 0; c < 192; ++c)
                        acc += (float)row[c] * fw[c * 9 + tap];
                }
            t[yo * 4 + xo] = acc;
        }
#pragma unroll
    for (int j = 0; j < 10; ++j) {
        float s = lb[j];
#pragma unroll
        for (int tt = 0; tt < 16; ++tt) s += t[tt] * lw[j * 16 + tt];
        scores[((long)f * BATCH + b) * 10 + j] = s;
    }
}

// ---------------------------------------------------------------------------
// loss: CE + mutual-exclusivity recurrence + stability; block-reduce + atomic
// ---------------------------------------------------------------------------
__global__ void zero_out(float* o) {
    if (blockIdx.x == 0 && threadIdx.x == 0) o[0] = 0.0f;
}

__global__ void loss_kernel(const float* __restrict__ scores, const int* __restrict__ y,
                            float* __restrict__ out) {
    __shared__ float red[256];
    const int b = blockIdx.x * 256 + threadIdx.x;
    float contrib = 0.0f;
    if (b < BATCH) {
        float sc[NFWD][10];
        for (int f = 0; f < NFWD; ++f)
            for (int c = 0; c < 10; ++c)
                sc[f][c] = scores[((long)f * BATCH + b) * 10 + c];
        const int lbl = y[b];
        float ce = 0.0f, me = 0.0f, st = 0.0f;
        for (int f = 0; f < NFWD; ++f) {
            float mx = sc[f][0];
            for (int c = 1; c < 10; ++c) mx = sc[f][c] > mx ? sc[f][c] : mx;
            float ex[10], se = 0.0f;
            for (int c = 0; c < 10; ++c) { ex[c] = __expf(sc[f][c] - mx); se += ex[c]; }
            ce += (__logf(se) + mx - sc[f][lbl]);
            float pp[10];
            for (int c = 0; c < 10; ++c) pp[c] = ex[c] / se;
            float s = 0.0f;
            for (int c = 0; c < 10; ++c) {
                float q = 1.0f;
                for (int j = 0; j < 10; ++j)
                    if (j != c) q *= (1.0f - pp[j]);
                s = (s + pp[c]) * q;
            }
            me += s;
        }
        for (int f1 = 0; f1 < NFWD; ++f1)
            for (int f2 = f1 + 1; f2 < NFWD; ++f2)
                for (int c = 0; c < 10; ++c) {
                    const float d = sc[f1][c] - sc[f2][c];
                    st += d * d;
                }
        const float inv = 1.0f / (float)(NFWD * BATCH);
        contrib = ce * inv - me * inv + st * inv;   // ME_W = ST_W = 1
    }
    red[threadIdx.x] = contrib;
    __syncthreads();
    for (int s = 128; s > 0; s >>= 1) {
        if (threadIdx.x < s) red[threadIdx.x] += red[threadIdx.x + s];
        __syncthreads();
    }
    if (threadIdx.x == 0) atomicAdd(out, red[0]);
}

// ---------------------------------------------------------------------------
// host side
// ---------------------------------------------------------------------------
static inline void launch_block(int NT, const _Float16* in, const _Float16* wkT,
                                const _Float16* wcT, const float* bconv,
                                const float* bcoup, const float* rz,
                                _Float16* out, int Cin, int H, int W,
                                hipStream_t stream) {
    const long Mtot = (long)BATCH * H * W;
    const int grid = (int)((Mtot + 127) / 128);
    switch (NT) {
        case 2: conv_block_gemm<2><<<grid, 256, 0, stream>>>(in, wkT, wcT, bconv, bcoup, rz, out, Cin, H, W); break;
        case 3: conv_block_gemm<3><<<grid, 256, 0, stream>>>(in, wkT, wcT, bconv, bcoup, rz, out, Cin, H, W); break;
        case 4: conv_block_gemm<4><<<grid, 256, 0, stream>>>(in, wkT, wcT, bconv, bcoup, rz, out, Cin, H, W); break;
        case 5: conv_block_gemm<5><<<grid, 256, 0, stream>>>(in, wkT, wcT, bconv, bcoup, rz, out, Cin, H, W); break;
        case 6: conv_block_gemm<6><<<grid, 256, 0, stream>>>(in, wkT, wcT, bconv, bcoup, rz, out, Cin, H, W); break;
    }
}

extern "C" void kernel_launch(void* const* d_in, const int* in_sizes, int n_in,
                              void* d_out, int out_size, void* d_ws, size_t ws_size,
                              hipStream_t stream) {
    (void)in_sizes; (void)n_in; (void)out_size; (void)ws_size;

    // inputs in setup_inputs() dict order
    const float* x       = (const float*)d_in[0];  // [1024,1,28,28]
    const int*   y       = (const int*)d_in[1];    // [1024]
    const float* conv0_w = (const float*)d_in[2];  // [32]
    const float* conv0_b = (const float*)d_in[3];  // [32]
    const float* final_w = (const float*)d_in[4];  // [1,192,3,3]
    const float* final_b = (const float*)d_in[5];  // [1]
    const float* lin_w   = (const float*)d_in[6];  // [10,16]
    const float* lin_b   = (const float*)d_in[7];  // [10]
    const float* blk_cw[5], *blk_cb[5], *blk_pw[5], *blk_pb[5], *blk_rz[5];
    for (int i = 0; i < 5; ++i) {
        blk_cw[i] = (const float*)d_in[8 + 5 * i + 0];
        blk_cb[i] = (const float*)d_in[8 + 5 * i + 1];
        blk_pw[i] = (const float*)d_in[8 + 5 * i + 2];
        blk_pb[i] = (const float*)d_in[8 + 5 * i + 3];
        blk_rz[i] = (const float*)d_in[8 + 5 * i + 4];
    }

    // workspace layout (bytes)
    char* ws = (char*)d_ws;
    float*     scores = (float*)ws;                                   // 4*1024*10 f32
    _Float16*  wkAll  = (_Float16*)(ws + 163840);                     // 645120 f16
    _Float16*  wcAll  = (_Float16*)(ws + 1454080);                    // 71680 f16
    const long MAXACT = (long)BATCH * 28 * 28 * 64;                   // 51,380,224 f16
    _Float16*  actA   = (_Float16*)(ws + 1597440);
    _Float16*  actB   = actA + MAXACT;

    static const long wkOff[5] = {0, 18432, 73728, 184320, 368640};
    static const long wcOff[5] = {0, 2048, 8192, 20480, 40960};
    static const int  Hin[5]   = {28, 19, 13, 9, 6};
    static const int  Hout[5]  = {19, 13, 9, 6, 4};

    zero_out<<<1, 32, 0, stream>>>((float*)d_out);

    // one-time weight repack (deterministic; weights are f-independent)
    for (int i = 0; i < 5; ++i) {
        const int Cin = 32 * (i + 1), Cout = 32 * (i + 2);
        {
            const long total = (long)Cin * 9 * Cout;
            repack_conv3x3<<<(int)((total + 255) / 256), 256, 0, stream>>>(
                blk_cw[i], wkAll + wkOff[i], Cin, Cout);
        }
        {
            const long total = (long)Cin * Cout;
            repack_conv1x1<<<(int)((total + 255) / 256), 256, 0, stream>>>(
                blk_pw[i], wcAll + wcOff[i], Cin, Cout);
        }
    }

    // four stochastic forward passes
    for (int f = 0; f < NFWD; ++f) {
        {
            const long total = (long)BATCH * 28 * 28;
            drop_conv0<<<(int)((total + 255) / 256), 256, 0, stream>>>(
                x, conv0_w, conv0_b, actA, f);
        }
        for (int i = 0; i < 5; ++i) {
            const int Cin = 32 * (i + 1), Cout = 32 * (i + 2);
            launch_block(i + 2, actA, wkAll + wkOff[i], wcAll + wcOff[i],
                         blk_cb[i], blk_pb[i], blk_rz[i], actB,
                         Cin, Hin[i], Hin[i], stream);
            const long ptotal = (long)BATCH * Hout[i] * Hout[i] * Cout;
            fpool_kernel<<<(int)((ptotal + 255) / 256), 256, 0, stream>>>(
                actB, actA, Hin[i], Hout[i], Cout);
        }
        final_linear<<<(BATCH + 63) / 64, 64, 0, stream>>>(
            actA, final_w, final_b, lin_w, lin_b, scores, f);
    }

    loss_kernel<<<(BATCH + 255) / 256, 256, 0, stream>>>(scores, y, (float*)d_out);
}
